// ExternalBidirectionalMambaLayer_47639777247253
// MI455X (gfx1250) — matrix operand order repaired and checked by reading, not compiled, not run on hardware
//
#include <hip/hip_runtime.h>
#include <hip/hip_bf16.h>

// ---------------------------------------------------------------------------
// Bidirectional Mamba for MI455X (gfx1250, wave32, WMMA).
// - GEMMs: v_wmma_f32_16x16x32_bf16, 32x64 tile per wave (8 WMMAs / K-step).
//   All GEMM operands are L2-resident (192 MB L2), so reuse is served by L2.
// - Selective scan: chunked parallel decomposition (16 chunks along L),
//   B/C staged into LDS via GLOBAL_LOAD_ASYNC_TO_LDS_B128 (ASYNCcnt).
// ---------------------------------------------------------------------------

typedef __attribute__((ext_vector_type(16))) __bf16 v16bf;
typedef __attribute__((ext_vector_type(8)))  float  v8f;
typedef __attribute__((ext_vector_type(4)))  int    v4i;

#define BSZ   4
#define SEQ   4096
#define DM    512
#define DI    1024
#define DS    16
#define DTR   32
#define NTOK  (BSZ * SEQ)
#define NCH   16               // scan chunks along L
#define CL    (SEQ / NCH)      // 256 steps per chunk

#if defined(__has_builtin)
#  if __has_builtin(__builtin_amdgcn_global_load_async_to_lds_b128)
#    define USE_ASYNC_LDS 1
#  endif
#endif

#ifdef USE_ASYNC_LDS
typedef __attribute__((address_space(1))) v4i* gv4_ptr;   // global (AS1) 16B vector
typedef __attribute__((address_space(3))) v4i* lv4_ptr;   // LDS (AS3) 16B vector
__device__ __forceinline__ void async_wait0() {
#if __has_builtin(__builtin_amdgcn_s_wait_asynccnt)
    __builtin_amdgcn_s_wait_asynccnt(0);
#else
    asm volatile("s_wait_asynccnt 0x0" ::: "memory");
#endif
}
#endif

__device__ __forceinline__ float fsigmoid(float x) { return 1.0f / (1.0f + __expf(-x)); }
__device__ __forceinline__ float fsilu(float x)    { return x * fsigmoid(x); }

// ------------------------- WMMA fragment helpers ---------------------------
// A-matrix 16x32 bf16: lanes 0-15 hold M=lane, K 0..15;
// lanes 16-31 hold M=lane-16, K 16..31 (same k order as B); 2 halves per VGPR.
__device__ __forceinline__ v16bf load_a_frag(const __bf16* __restrict__ A,
                                             int row_base, int k_base, int lda) {
    int lane  = threadIdx.x & 31;
    int m     = lane & 15;
    int khalf = (lane >> 4) ? 16 : 0;
    const __bf16* p = A + (size_t)(row_base + m) * lda + k_base;
    v16bf a;
#pragma unroll
    for (int j = 0; j < 16; ++j) {
        int k = khalf + j;
        a[j] = p[k];
    }
    return a;
}

// B-matrix 32x16 bf16: lane holds column N=lane&15; lanes 0-15 carry K=0..15,
// lanes 16-31 carry K=16..31 (weights stored row-major [N x K], B = W^T).
__device__ __forceinline__ v16bf load_b_frag(const __bf16* __restrict__ W,
                                             int n_base, int k_base, int ldw) {
    int lane = threadIdx.x & 31;
    int n    = lane & 15;
    int kh   = (lane >> 4) ? 16 : 0;
    const __bf16* p = W + (size_t)(n_base + n) * ldw + k_base + kh;
    v16bf b;
#pragma unroll
    for (int j = 0; j < 16; ++j) b[j] = p[j];
    return b;
}

// One wave computes a 32x64 output tile: 2 A frags + 4 B frags -> 8 WMMAs/K-step.
__device__ __forceinline__ void wmma_tile_32x64(const __bf16* __restrict__ A,
                                                const __bf16* __restrict__ W,
                                                int K, int row, int col, v8f acc[2][4]) {
    for (int k = 0; k < K; k += 32) {
        v16bf a0 = load_a_frag(A, row,      k, K);
        v16bf a1 = load_a_frag(A, row + 16, k, K);
#pragma unroll
        for (int i = 0; i < 4; ++i) {
            v16bf b = load_b_frag(W, col + 16 * i, k, K);
            acc[0][i] = __builtin_amdgcn_wmma_f32_16x16x32_bf16(
                false, a0, false, b, (short)0, acc[0][i], false, false);
            acc[1][i] = __builtin_amdgcn_wmma_f32_16x16x32_bf16(
                false, a1, false, b, (short)0, acc[1][i], false, false);
        }
    }
}

// ------------------------------ GEMM kernels -------------------------------
// C/D layout (ISA): VGPR r -> M = r + 8*(lane>=16), N = lane&15.
__global__ void k_gemm_f32(const __bf16* __restrict__ A, const __bf16* __restrict__ W,
                           float* __restrict__ C, int M, int N, int K, int ldc) {
    int wave    = (blockIdx.x * blockDim.x + threadIdx.x) >> 5;
    int tiles_n = N >> 6;
    int tiles   = (M >> 5) * tiles_n;
    if (wave >= tiles) return;                 // wave-uniform: EXEC stays all-ones
    int row = (wave / tiles_n) << 5;
    int col = (wave % tiles_n) << 6;
    v8f acc[2][4] = {};
    wmma_tile_32x64(A, W, K, row, col, acc);
    int lane = threadIdx.x & 31;
    int n    = lane & 15;
    int mo   = (lane >> 4) << 3;
#pragma unroll
    for (int ms = 0; ms < 2; ++ms)
#pragma unroll
        for (int i = 0; i < 4; ++i)
#pragma unroll
            for (int r = 0; r < 8; ++r)
                C[(size_t)(row + 16 * ms + mo + r) * ldc + col + 16 * i + n] = acc[ms][i][r];
}

__global__ void k_gemm_bias_f32(const __bf16* __restrict__ A, const __bf16* __restrict__ W,
                                const float* __restrict__ bias, float* __restrict__ C,
                                int M, int N, int K, int ldc) {
    int wave    = (blockIdx.x * blockDim.x + threadIdx.x) >> 5;
    int tiles_n = N >> 6;
    int tiles   = (M >> 5) * tiles_n;
    if (wave >= tiles) return;
    int row = (wave / tiles_n) << 5;
    int col = (wave % tiles_n) << 6;
    v8f acc[2][4] = {};
    wmma_tile_32x64(A, W, K, row, col, acc);
    int lane = threadIdx.x & 31;
    int n    = lane & 15;
    int mo   = (lane >> 4) << 3;
#pragma unroll
    for (int ms = 0; ms < 2; ++ms)
#pragma unroll
        for (int i = 0; i < 4; ++i) {
            float bv = bias[col + 16 * i + n];
#pragma unroll
            for (int r = 0; r < 8; ++r)
                C[(size_t)(row + 16 * ms + mo + r) * ldc + col + 16 * i + n] =
                    acc[ms][i][r] + bv;
        }
}

// Store bf16 into the concat buffer; optional time reversal for the backward dir.
__global__ void k_gemm_store_bf16(const __bf16* __restrict__ A, const __bf16* __restrict__ W,
                                  __bf16* __restrict__ Cdst, int M, int N, int K,
                                  int ldc, int col_off, int do_rev) {
    int wave    = (blockIdx.x * blockDim.x + threadIdx.x) >> 5;
    int tiles_n = N >> 6;
    int tiles   = (M >> 5) * tiles_n;
    if (wave >= tiles) return;
    int row = (wave / tiles_n) << 5;
    int col = (wave % tiles_n) << 6;
    v8f acc[2][4] = {};
    wmma_tile_32x64(A, W, K, row, col, acc);
    int lane = threadIdx.x & 31;
    int n    = lane & 15;
    int mo   = (lane >> 4) << 3;
#pragma unroll
    for (int ms = 0; ms < 2; ++ms)
#pragma unroll
        for (int i = 0; i < 4; ++i)
#pragma unroll
            for (int r = 0; r < 8; ++r) {
                int tok = row + 16 * ms + mo + r;
                int dst = tok;
                if (do_rev) {
                    int b = tok >> 12, t = tok & (SEQ - 1);
                    dst = (b << 12) + (SEQ - 1 - t);
                }
                Cdst[(size_t)dst * ldc + col_off + col + 16 * i + n] = (__bf16)acc[ms][i][r];
            }
}

// ---------------------------- conversion kernels ---------------------------
__global__ void k_cvt_bf16(const float* __restrict__ src, __bf16* __restrict__ dst, int n) {
    int i = blockIdx.x * blockDim.x + threadIdx.x;
    if (i < n) dst[i] = (__bf16)src[i];
}

// bf16 copy of x plus a time-reversed bf16 copy (for the backward direction).
__global__ void k_x_dual(const float* __restrict__ x, __bf16* __restrict__ xbf,
                         __bf16* __restrict__ xrev) {
    int i = blockIdx.x * blockDim.x + threadIdx.x;
    if (i >= NTOK * DM) return;
    int c = i % DM, tok = i / DM;
    int t = tok & (SEQ - 1), b = tok >> 12;
    float v = x[i];
    xbf[i] = (__bf16)v;
    xrev[(size_t)((b << 12) + (SEQ - 1 - t)) * DM + c] = (__bf16)v;
}

// ------------------------- conv + silu (depthwise 4) -----------------------
__global__ void k_conv(const float* __restrict__ xz, const float* __restrict__ cw,
                       const float* __restrict__ cb, float* __restrict__ xi,
                       __bf16* __restrict__ xibf) {
    int i = blockIdx.x * blockDim.x + threadIdx.x;
    if (i >= NTOK * DI) return;
    int d = i & (DI - 1), tok = i >> 10;
    int t = tok & (SEQ - 1);
    float acc = cb[d];
#pragma unroll
    for (int k = 0; k < 4; ++k) {
        int tt = t - 3 + k;
        if (tt >= 0) acc += cw[d * 4 + k] * xz[(size_t)(tok - 3 + k) * (2 * DI) + d];
    }
    float s = fsilu(acc);
    xi[i]   = s;
    xibf[i] = (__bf16)s;
}

// ------------------------- dt = softplus(dbc@dtW + b) ----------------------
__global__ void k_dtproj(const float* __restrict__ dbc, const float* __restrict__ dt_w,
                         const float* __restrict__ dt_b, float* __restrict__ dt) {
    int i = blockIdx.x * blockDim.x + threadIdx.x;
    if (i >= NTOK * DI) return;
    int d = i & (DI - 1), tok = i >> 10;
    float acc = dt_b[d];
    const float* r = dbc + (size_t)tok * 64;
    const float* w = dt_w + d * DTR;
#pragma unroll
    for (int j = 0; j < DTR; ++j) acc += r[j] * w[j];
    dt[i] = (acc > 20.f) ? acc : log1pf(__expf(acc));
}

// --------------------- chunked selective scan: phase 1 ---------------------
// Each block: (b, chunk c, 128 channels). Local scan with h0=0; records
// chunk-final state and the chunk's total dt (decay exponent = exp(A*Σdt)).
__global__ void k_scan_local(const float* __restrict__ xi, const float* __restrict__ dt,
                             const float* __restrict__ dbc, const float* __restrict__ A_log,
                             float* __restrict__ ys, float* __restrict__ hstate,
                             float* __restrict__ dtsum) {
    __shared__ float BC[64][32];               // B (cols 0..15) and C (16..31)
    int bid  = blockIdx.x;
    int dgrp = bid & 7;
    int c    = (bid >> 3) & (NCH - 1);
    int b    = bid >> 7;
    int d    = dgrp * 128 + threadIdx.x;
    float Areg[DS], h[DS];
#pragma unroll
    for (int s = 0; s < DS; ++s) { Areg[s] = -__expf(A_log[d * DS + s]); h[s] = 0.f; }
    float dts = 0.f;
    int t0 = c * CL;
    for (int tc = 0; tc < CL; tc += 64) {
        __syncthreads();                       // protect prior-iteration readers
#ifdef USE_ASYNC_LDS
        // 64 rows x 32 floats = 512 x 16B chunks; 4 async B128 copies/thread
#pragma unroll
        for (int q = 0; q < 4; ++q) {
            int e  = threadIdx.x + 128 * q;
            int rr = e >> 3, pp = e & 7;       // row, 16B-part within row
            const float* gsrc = dbc + (size_t)(b * SEQ + t0 + tc + rr) * 64 + 32 + pp * 4;
            __builtin_amdgcn_global_load_async_to_lds_b128(
                (gv4_ptr)gsrc, (lv4_ptr)&BC[rr][pp * 4], 0, 0);
        }
        async_wait0();
#else
#pragma unroll
        for (int q = 0; q < 16; ++q) {
            int e = threadIdx.x + 128 * q;     // 2048 floats = 64 steps x 32
            int rr = e >> 5, cc = e & 31;
            BC[rr][cc] = dbc[(size_t)(b * SEQ + t0 + tc + rr) * 64 + 32 + cc];
        }
#endif
        __syncthreads();
        for (int i = 0; i < 64; ++i) {
            size_t g = (size_t)(b * SEQ + t0 + tc + i) * DI + d;
            float dtv = dt[g], xv = xi[g];
            dts += dtv;
            float yv = 0.f, dx = dtv * xv;
#pragma unroll
            for (int s = 0; s < DS; ++s) {
                h[s] = __expf(dtv * Areg[s]) * h[s] + dx * BC[i][s];
                yv  += h[s] * BC[i][16 + s];
            }
            ys[g] = yv;
        }
    }
    size_t hi = (size_t)((b * NCH + c) * DI + d);
#pragma unroll
    for (int s = 0; s < DS; ++s) hstate[hi * DS + s] = h[s];
    dtsum[hi] = dts;
}

// --------------------- phase 2: tiny serial carry combine ------------------
// Replaces hstate (chunk-final) with hinit (state entering each chunk) in place.
__global__ void k_scan_carry(const float* __restrict__ A_log, const float* __restrict__ dtsum,
                             float* __restrict__ hstate) {
    int i = blockIdx.x * blockDim.x + threadIdx.x;
    if (i >= BSZ * DI) return;
    int d = i & (DI - 1), b = i >> 10;
    float Areg[DS], carry[DS];
#pragma unroll
    for (int s = 0; s < DS; ++s) { Areg[s] = -__expf(A_log[d * DS + s]); carry[s] = 0.f; }
    for (int c = 0; c < NCH; ++c) {
        size_t hi = (size_t)((b * NCH + c) * DI + d);
        float dsv = dtsum[hi];
#pragma unroll
        for (int s = 0; s < DS; ++s) {
            float tmp = hstate[hi * DS + s];
            hstate[hi * DS + s] = carry[s];
            carry[s] = carry[s] * __expf(Areg[s] * dsv) + tmp;
        }
    }
}

// ------- phase 3: apply carried-state correction + gate, emit bf16 ---------
__global__ void k_scan_fix(const float* __restrict__ xi, const float* __restrict__ dt,
                           const float* __restrict__ dbc, const float* __restrict__ A_log,
                           const float* __restrict__ Dp, const float* __restrict__ xz,
                           const float* __restrict__ ys, const float* __restrict__ hstate,
                           __bf16* __restrict__ ybf) {
    __shared__ float Csh[64][16];
    int bid  = blockIdx.x;
    int dgrp = bid & 7;
    int c    = (bid >> 3) & (NCH - 1);
    int b    = bid >> 7;
    int d    = dgrp * 128 + threadIdx.x;
    float Areg[DS], hc[DS];
    size_t hi = (size_t)((b * NCH + c) * DI + d);
#pragma unroll
    for (int s = 0; s < DS; ++s) {
        Areg[s] = -__expf(A_log[d * DS + s]);
        hc[s]   = hstate[hi * DS + s];
    }
    float Dv = Dp[d];
    int t0 = c * CL;
    for (int tc = 0; tc < CL; tc += 64) {
        __syncthreads();
#ifdef USE_ASYNC_LDS
        // 64 rows x 16 floats = 256 x 16B chunks; 2 async B128 copies/thread
#pragma unroll
        for (int q = 0; q < 2; ++q) {
            int e  = threadIdx.x + 128 * q;
            int rr = e >> 2, pp = e & 3;
            const float* gsrc = dbc + (size_t)(b * SEQ + t0 + tc + rr) * 64 + 48 + pp * 4;
            __builtin_amdgcn_global_load_async_to_lds_b128(
                (gv4_ptr)gsrc, (lv4_ptr)&Csh[rr][pp * 4], 0, 0);
        }
        async_wait0();
#else
#pragma unroll
        for (int q = 0; q < 8; ++q) {
            int e = threadIdx.x + 128 * q;     // 1024 floats = 64 steps x 16 (C only)
            int rr = e >> 4, cc = e & 15;
            Csh[rr][cc] = dbc[(size_t)(b * SEQ + t0 + tc + rr) * 64 + 48 + cc];
        }
#endif
        __syncthreads();
        for (int i = 0; i < 64; ++i) {
            size_t g = (size_t)(b * SEQ + t0 + tc + i) * DI + d;
            float dtv = dt[g];
            float ycorr = 0.f;
#pragma unroll
            for (int s = 0; s < DS; ++s) {
                hc[s] *= __expf(dtv * Areg[s]);
                ycorr += hc[s] * Csh[i][s];
            }
            float xv = xi[g];
            float zv = xz[(size_t)(b * SEQ + t0 + tc + i) * (2 * DI) + DI + d];
            float yv = (ys[g] + ycorr + xv * Dv) * fsilu(zv);
            ybf[g] = (__bf16)yv;
        }
    }
}

// ------------------------------- launcher ----------------------------------
static inline int ew_blocks(long long n) { return (int)((n + 255) / 256); }
static inline int gemm_blocks(int M, int N) {
    int tiles = (M >> 5) * (N >> 6);           // one wave per 32x64 tile, 8 waves/block
    return (tiles + 7) / 8;
}

extern "C" void kernel_launch(void* const* d_in, const int* in_sizes, int n_in,
                              void* d_out, int out_size, void* d_ws, size_t ws_size,
                              hipStream_t stream) {
    const float* x = (const float*)d_in[0];
    const float* dir_in_w[2]   = { (const float*)d_in[1],  (const float*)d_in[10] };
    const float* dir_conv_w[2] = { (const float*)d_in[2],  (const float*)d_in[11] };
    const float* dir_conv_b[2] = { (const float*)d_in[3],  (const float*)d_in[12] };
    const float* dir_xproj[2]  = { (const float*)d_in[4],  (const float*)d_in[13] };
    const float* dir_dt_w[2]   = { (const float*)d_in[5],  (const float*)d_in[14] };
    const float* dir_dt_b[2]   = { (const float*)d_in[6],  (const float*)d_in[15] };
    const float* dir_A_log[2]  = { (const float*)d_in[7],  (const float*)d_in[16] };
    const float* dir_D[2]      = { (const float*)d_in[8],  (const float*)d_in[17] };
    const float* dir_out_w[2]  = { (const float*)d_in[9],  (const float*)d_in[18] };
    const float* fin_w = (const float*)d_in[19];
    const float* fin_b = (const float*)d_in[20];
    float* out = (float*)d_out;

    // ---- carve workspace ----
    char* p = (char*)d_ws;
    auto carve = [&](size_t bytes) -> void* {
        void* r = (void*)p;
        p += (bytes + 255) & ~(size_t)255;
        return r;
    };
    __bf16* xbf   = (__bf16*)carve((size_t)NTOK * DM * 2);
    __bf16* xrev  = (__bf16*)carve((size_t)NTOK * DM * 2);
    __bf16* w_in  = (__bf16*)carve((size_t)2 * DI * DM * 2);
    __bf16* w_xp  = (__bf16*)carve((size_t)(DTR + 2 * DS) * DI * 2);
    __bf16* w_out = (__bf16*)carve((size_t)DM * DI * 2);
    __bf16* w_fin = (__bf16*)carve((size_t)DM * (2 * DM) * 2);
    float*  xz    = (float*) carve((size_t)NTOK * 2 * DI * 4);
    float*  xiF   = (float*) carve((size_t)NTOK * DI * 4);
    __bf16* xiB   = (__bf16*)carve((size_t)NTOK * DI * 2);
    float*  dbc   = (float*) carve((size_t)NTOK * 64 * 4);
    float*  dtb   = (float*) carve((size_t)NTOK * DI * 4);
    float*  ysb   = (float*) carve((size_t)NTOK * DI * 4);
    float*  hst   = (float*) carve((size_t)BSZ * NCH * DI * DS * 4);
    float*  dts   = (float*) carve((size_t)BSZ * NCH * DI * 4);
    __bf16* ybf   = (__bf16*)carve((size_t)NTOK * DI * 2);
    __bf16* fb    = (__bf16*)carve((size_t)NTOK * 2 * DM * 2);

    // ---- x -> bf16 (+ time-reversed copy), final weight -> bf16 ----
    k_x_dual<<<ew_blocks((long long)NTOK * DM), 256, 0, stream>>>(x, xbf, xrev);
    k_cvt_bf16<<<ew_blocks(DM * 2 * DM), 256, 0, stream>>>(fin_w, w_fin, DM * 2 * DM);

    for (int dir = 0; dir < 2; ++dir) {
        const __bf16* xin = dir ? xrev : xbf;
        // per-direction weight conversion (scratch reused; stream-ordered)
        k_cvt_bf16<<<ew_blocks(2 * DI * DM), 256, 0, stream>>>(dir_in_w[dir], w_in, 2 * DI * DM);
        k_cvt_bf16<<<ew_blocks((DTR + 2 * DS) * DI), 256, 0, stream>>>(dir_xproj[dir], w_xp,
                                                                       (DTR + 2 * DS) * DI);
        k_cvt_bf16<<<ew_blocks(DM * DI), 256, 0, stream>>>(dir_out_w[dir], w_out, DM * DI);

        // in_proj: (NTOK x 512) @ (2048 x 512)^T -> xz (NTOK x 2048)
        k_gemm_f32<<<gemm_blocks(NTOK, 2 * DI), 256, 0, stream>>>(
            xin, w_in, xz, NTOK, 2 * DI, DM, 2 * DI);
        // depthwise conv + silu
        k_conv<<<ew_blocks((long long)NTOK * DI), 256, 0, stream>>>(
            xz, dir_conv_w[dir], dir_conv_b[dir], xiF, xiB);
        // x_proj: (NTOK x 1024) @ (64 x 1024)^T -> dbc (NTOK x 64)
        k_gemm_f32<<<gemm_blocks(NTOK, 64), 256, 0, stream>>>(
            xiB, w_xp, dbc, NTOK, 64, DI, 64);
        // dt projection + softplus
        k_dtproj<<<ew_blocks((long long)NTOK * DI), 256, 0, stream>>>(
            dbc, dir_dt_w[dir], dir_dt_b[dir], dtb);
        // chunked selective scan (3 phases) + fused gating -> ybf (bf16)
        k_scan_local<<<BSZ * NCH * (DI / 128), 128, 0, stream>>>(
            xiF, dtb, dbc, dir_A_log[dir], ysb, hst, dts);
        k_scan_carry<<<(BSZ * DI + 255) / 256, 256, 0, stream>>>(dir_A_log[dir], dts, hst);
        k_scan_fix<<<BSZ * NCH * (DI / 128), 128, 0, stream>>>(
            xiF, dtb, dbc, dir_A_log[dir], dir_D[dir], xz, ysb, hst, ybf);
        // out_proj -> concat buffer (bf16), backward dir written time-reversed
        k_gemm_store_bf16<<<gemm_blocks(NTOK, DM), 256, 0, stream>>>(
            ybf, w_out, fb, NTOK, DM, DI, 2 * DM, dir ? DM : 0, dir);
    }

    // final: (NTOK x 1024) @ (512 x 1024)^T + bias -> out (fp32)
    k_gemm_bias_f32<<<gemm_blocks(NTOK, DM), 256, 0, stream>>>(
        fb, w_fin, fin_b, out, NTOK, DM, 2 * DM, DM);
    (void)in_sizes; (void)n_in; (void)out_size; (void)ws_size;
}
